// Conv1d_27522150432984
// MI455X (gfx1250) — compile-verified, matrix operand order checked by reading, not run on hardware
//
#include <hip/hip_runtime.h>

typedef __bf16 bf16_t;
typedef __attribute__((ext_vector_type(16))) __bf16 v16bf;
typedef __attribute__((ext_vector_type(8)))  __bf16 v8bf;
typedef __attribute__((ext_vector_type(8)))  float  v8f;

constexpr int CIN   = 256;
constexpr int COUT  = 256;
constexpr int WLEN  = 4096;
constexpr int BATCH = 32;
constexpr int KKTOT = CIN * 3;       // 768, permuted K = u*256 + ci
constexpr int LDPF  = KKTOT + 8;     // A row stride: 388 dwords = 4 mod 64 banks
constexpr int CIH   = 128;           // ci per K-half (B slab restaged per half)
constexpr int NPOS  = 258;           // x slab positions: [base-1, base+256]
constexpr int CSTRH = CIH + 8;       // B slab row stride: 68 dwords = 4 mod 64 banks

__device__ __forceinline__ v16bf ld_frag(const bf16_t* p) {
    v8bf a = *(const v8bf*)p;
    v8bf b = *(const v8bf*)(p + 16);
    return __builtin_shufflevector(a, b, 0,1,2,3,4,5,6,7,8,9,10,11,12,13,14,15);
}

__global__ __launch_bounds__(256)
void conv1d_wmma_bf16x3(const float* __restrict__ x,
                        const float* __restrict__ weight,
                        const float* __restrict__ bias,
                        float* __restrict__ out)
{
    __shared__ __align__(16) bf16_t smA_hi[32 * LDPF];    // 49.7 KB, full K
    __shared__ __align__(16) bf16_t smA_lo[32 * LDPF];    // 49.7 KB
    __shared__ __align__(16) bf16_t smB_hi[NPOS * CSTRH]; // 70.2 KB, [pos][ci_local]
    __shared__ __align__(16) bf16_t smB_lo[NPOS * CSTRH]; // 70.2 KB   -> 239.7 KB total

    const int tid  = threadIdx.x;
    const int mt   = blockIdx.x;      // 32-row C_out tile: 0..7
    const int nb   = blockIdx.y;      // 256-column span: 0..511
    const int wave = tid >> 5;        // owns cols [wave*32, wave*32+32)
    const int lane = tid & 31;
    const int half = lane >> 4;
    const int m    = lane & 15;

    const int nbase = nb * 256;            // spans never cross batch boundary
    const int bb    = nbase >> 12;
    const int ibase = nbase & (WLEN - 1);
    const float* xrow = x + (size_t)bb * CIN * WLEN;

    // ---- Stage A once: 32 rows x 768 K (u-major), bf16 hi/lo. 32 iters x 3 loads ----
    #pragma unroll 4
    for (int idx = tid; idx < 32 * CIN; idx += 256) {
        int r  = idx >> 8;
        int ci = idx & 255;
        const float* wp = weight + (size_t)(mt * 32 + r) * KKTOT + ci * 3;
        #pragma unroll
        for (int uu = 0; uu < 3; ++uu) {
            float wv = wp[uu];
            bf16_t hv = (bf16_t)wv;
            smA_hi[r * LDPF + uu * 256 + ci] = hv;
            smA_lo[r * LDPF + uu * 256 + ci] = (bf16_t)(wv - (float)hv);
        }
    }

    v8f acc[2][2] = {};   // [mtile r][ntile s]

    for (int h = 0; h < 2; ++h) {          // K-halves: ci in [h*128, h*128+128)
        if (h) __syncthreads();            // previous compute done before restage

        // ---- Stage B interior: p in [1,257) is always in-bounds -> float4 loads.
        //      128 ci x 64 quads = 8192 quads, 32 iters/thread, fully coalesced ----
        #pragma unroll 4
        for (int idx = tid; idx < CIH * 64; idx += 256) {
            int cil = idx >> 6;
            int q   = idx & 63;
            const float4 v4 = *(const float4*)(xrow + (size_t)(h * CIH + cil) * WLEN
                                               + ibase + q * 4);
            const float vv[4] = {v4.x, v4.y, v4.z, v4.w};
            #pragma unroll
            for (int k = 0; k < 4; ++k) {
                float v = vv[k];
                bf16_t hv = (bf16_t)v;
                int p = 1 + q * 4 + k;
                smB_hi[p * CSTRH + cil] = hv;
                smB_lo[p * CSTRH + cil] = (bf16_t)(v - (float)hv);
            }
        }
        // ---- Stage B edges: p=0 and p=257, one element per thread, clamp+zero ----
        {
            int e   = tid >> 7;            // 0 or 1
            int cil = tid & 127;
            int p   = e ? 257 : 0;
            int gpos  = ibase - 1 + p;
            bool inb  = (unsigned)gpos < (unsigned)WLEN;
            int gposc = gpos < 0 ? 0 : (gpos > WLEN - 1 ? WLEN - 1 : gpos);
            float v = xrow[(size_t)(h * CIH + cil) * WLEN + gposc];
            v = inb ? v : 0.0f;
            bf16_t hv = (bf16_t)v;
            smB_hi[p * CSTRH + cil] = hv;
            smB_lo[p * CSTRH + cil] = (bf16_t)(v - (float)hv);
        }
        __syncthreads();

        // ---- Hot loop: 12 chunks x (16 ds_load_b128 -> 12 WMMA), 2Mx2N tile ----
        #pragma unroll
        for (int u = 0; u < 3; ++u) {
            #pragma unroll
            for (int c2 = 0; c2 < 4; ++c2) {
                const int ci0 = c2 * 32 + half * 8;
                const int kA  = u * 256 + h * CIH + ci0;

                v16bf a_hi[2], a_lo[2], b_hi[2], b_lo[2];
                #pragma unroll
                for (int r = 0; r < 2; ++r) {
                    const int rowA = (r * 16 + m) * LDPF + kA;
                    a_hi[r] = ld_frag(&smA_hi[rowA]);
                    a_lo[r] = ld_frag(&smA_lo[rowA]);
                }
                #pragma unroll
                for (int s = 0; s < 2; ++s) {
                    const int rowB = (wave * 32 + s * 16 + m + u) * CSTRH + ci0;
                    b_hi[s] = ld_frag(&smB_hi[rowB]);
                    b_lo[s] = ld_frag(&smB_lo[rowB]);
                }

                #pragma unroll
                for (int r = 0; r < 2; ++r)
                    #pragma unroll
                    for (int s = 0; s < 2; ++s) {
                        // bf16x3 fp32 emulation: hi*hi + hi*lo + lo*hi
                        acc[r][s] = __builtin_amdgcn_wmma_f32_16x16x32_bf16(
                            false, a_hi[r], false, b_hi[s], (short)0, acc[r][s], false, false);
                        acc[r][s] = __builtin_amdgcn_wmma_f32_16x16x32_bf16(
                            false, a_hi[r], false, b_lo[s], (short)0, acc[r][s], false, false);
                        acc[r][s] = __builtin_amdgcn_wmma_f32_16x16x32_bf16(
                            false, a_lo[r], false, b_hi[s], (short)0, acc[r][s], false, false);
                    }
            }
        }
    }

    // ---- Epilogue: bias + store (C/D layout: VGPR j -> M = half*8 + j, col = m) ----
    #pragma unroll
    for (int r = 0; r < 2; ++r) {
        const int co0 = mt * 32 + r * 16 + half * 8;
        v8f bv = *(const v8f*)(bias + co0);           // multiple of 8 -> 32B aligned
        #pragma unroll
        for (int s = 0; s < 2; ++s) {
            const int ii = ibase + wave * 32 + s * 16 + m;
            #pragma unroll
            for (int j = 0; j < 8; ++j) {
                float ov = acc[r][s][j] + bv[j];
                out[((size_t)bb * COUT + co0 + j) * WLEN + ii] = ov;
            }
        }
    }
}

extern "C" void kernel_launch(void* const* d_in, const int* in_sizes, int n_in,
                              void* d_out, int out_size, void* d_ws, size_t ws_size,
                              hipStream_t stream) {
    const float* x = (const float*)d_in[0];
    const float* w = (const float*)d_in[1];
    const float* b = (const float*)d_in[2];
    float* out = (float*)d_out;

    dim3 grid(COUT / 32, (BATCH * WLEN) / 256);  // (8, 512)
    conv1d_wmma_bf16x3<<<grid, dim3(256), 0, stream>>>(x, w, b, out);
}
